// RNN_14242111554072
// MI455X (gfx1250) — matrix operand order repaired and checked by reading, not compiled, not run on hardware
//
#include <hip/hip_runtime.h>
#include <hip/hip_bf16.h>
#include <math.h>

// ---------------------------------------------------------------------------
// RNN: h_t = tanh(x_t @ W_xh^T + h @ W_hh^T + b_hh); out_t = tanh(h_t @ W_hx^T + b_hx)
// Phase 1: Z = x @ W_xh^T + b_hh        (one big bf16-WMMA GEMM, all timesteps)
// Phase 2: persistent-kernel recurrence (512 steps, grid barrier between steps)
// Phase 3: out = tanh(H @ W_hx^T + b_hx) (one big bf16-WMMA GEMM)
// WMMA K-loops are software-pipelined (double-buffered fragments) so the
// per-wave s_wait_loadcnt before each WMMA is partial, not a full drain.
// ---------------------------------------------------------------------------

typedef __bf16 bf16;
typedef __attribute__((ext_vector_type(16))) __bf16 v16bf;
typedef __attribute__((ext_vector_type(8)))  __bf16 v8bf;
typedef __attribute__((ext_vector_type(8)))  float  v8f;

#define IN_F  512
#define HID   1024
#define BATCH 64
#define SEQ   512

// ----------------------------- f32 -> bf16 conversion -----------------------
__global__ void cvt_f32_bf16(const float* __restrict__ src, bf16* __restrict__ dst, int n) {
  int i = blockIdx.x * blockDim.x + threadIdx.x;
  if (i < n) dst[i] = (bf16)src[i];
}

__global__ void init_u32(unsigned* p) { *p = 0u; }

// ----------------------------- WMMA 16x(16*NB) strip over K -----------------
// A: [M,K] bf16 row-major. Weight W: [N,K] bf16 row-major (B operand = W^T,
// fetched as contiguous rows of W). One A-fragment is reused across NB
// B-fragments (register blocking along N). Double-buffered over k-steps:
// frags for step s+1 are issued before the WMMAs of step s, so each WMMA
// waits only for its own loads while the next step's loads are in flight.
template <int NB, int K>
__device__ __forceinline__ void wmma_strip(const bf16* __restrict__ A, int lda,
                                           const bf16* __restrict__ W, int ldb,
                                           int m0, int n0, int lane,
                                           v8f acc[NB]) {
  const int l  = lane & 15;
  const int hi = lane >> 4;
  const bf16* arow = A + (size_t)(m0 + l) * lda;
  const bf16* brow[NB];
#pragma unroll
  for (int j = 0; j < NB; ++j) brow[j] = W + (size_t)(n0 + j * 16 + l) * ldb;

  const v8f zero = {0.f, 0.f, 0.f, 0.f, 0.f, 0.f, 0.f, 0.f};
#pragma unroll
  for (int j = 0; j < NB; ++j) acc[j] = zero;

  // A 16x32 bf16 layout: lane(l,hi) = row m0+l, K = k+hi*16+[0..15]
  // (two contiguous 16B loads).
  auto loadA = [&](int k) -> v16bf {
    v8bf alo = *(const v8bf*)(arow + k + hi * 16);
    v8bf ahi = *(const v8bf*)(arow + k + hi * 16 + 8);
    v16bf a;
#pragma unroll
    for (int i = 0; i < 8; ++i) { a[i] = alo[i]; a[8 + i] = ahi[i]; }
    return a;
  };
  // B 32x16 bf16 layout: lane(l,hi) = column n, K = k+hi*16+[0..15]
  // (one contiguous 32B load of a W row).
  auto loadB = [&](const bf16* base, int k) -> v16bf {
    return *(const v16bf*)(base + k + hi * 16);
  };

  constexpr int STEPS = K / 32;
  static_assert(STEPS % 2 == 0, "K/32 must be even for the ping-pong pipeline");

  v16bf a0, a1, b0[NB], b1[NB];
  a0 = loadA(0);
#pragma unroll
  for (int j = 0; j < NB; ++j) b0[j] = loadB(brow[j], 0);

#pragma unroll
  for (int s = 1; s < STEPS; s += 2) {
    // issue loads for step s, then compute step s-1 (buffer 0)
    a1 = loadA(s * 32);
#pragma unroll
    for (int j = 0; j < NB; ++j) b1[j] = loadB(brow[j], s * 32);
#pragma unroll
    for (int j = 0; j < NB; ++j)
      acc[j] = __builtin_amdgcn_wmma_f32_16x16x32_bf16(
          false, a0, false, b0[j], (short)0, acc[j], false, false);
    // issue loads for step s+1 (if any), then compute step s (buffer 1)
    if (s + 1 < STEPS) {
      a0 = loadA((s + 1) * 32);
#pragma unroll
      for (int j = 0; j < NB; ++j) b0[j] = loadB(brow[j], (s + 1) * 32);
    }
#pragma unroll
    for (int j = 0; j < NB; ++j)
      acc[j] = __builtin_amdgcn_wmma_f32_16x16x32_bf16(
          false, a1, false, b1[j], (short)0, acc[j], false, false);
  }
}

// ----------------------------- Phase 1: Z = x @ W_xh^T + b_hh ---------------
// Xb: [B*SEQ, IN_F] bf16 (row m = b*SEQ+t), Wxh: [HID, IN_F] bf16
// Z:  [SEQ, B, HID] f32 (time-major so each step reads a contiguous slab)
__global__ void gemm_xWxh(const bf16* __restrict__ Xb, const bf16* __restrict__ Wxh,
                          const float* __restrict__ b_hh, float* __restrict__ Z) {
  const int wave = (blockIdx.x * blockDim.x + threadIdx.x) >> 5;
  const int lane = threadIdx.x & 31;
  const int nstrips = HID / 64;               // 16 strips of 64 along N
  const int m0 = (wave / nstrips) * 16;
  const int n0 = (wave % nstrips) * 64;
  v8f acc[4];
  wmma_strip<4, IN_F>(Xb, IN_F, Wxh, IN_F, m0, n0, lane, acc);
  const int l = lane & 15, hi = lane >> 4;
#pragma unroll
  for (int j = 0; j < 4; ++j) {
    const float bias = b_hh[n0 + j * 16 + l];
#pragma unroll
    for (int r = 0; r < 8; ++r) {
      int m = m0 + hi * 8 + r;                // m = b*SEQ + t
      int b = m / SEQ, t = m % SEQ;
      Z[(size_t)t * BATCH * HID + (size_t)b * HID + n0 + j * 16 + l] = acc[j][r] + bias;
    }
  }
}

// ----------------------------- Phase 2: persistent recurrence ---------------
// 32 blocks x 256 threads = 256 waves; one 16x16 tile per wave per step
// (max parallelism in the latency-critical serial chain). Grid-wide barrier
// between steps via a monotonic generation counter with agent-scope acq/rel
// atomics (release writes back WGP$, acquire invalidates).
__global__ void rnn_recurrence(const bf16* __restrict__ h0b,
                               const bf16* __restrict__ Whh,
                               const float* __restrict__ Z,
                               bf16* __restrict__ Hb16,
                               float* __restrict__ hF32,
                               unsigned* __restrict__ barrier) {
  const int wave = (blockIdx.x * blockDim.x + threadIdx.x) >> 5;
  const int lane = threadIdx.x & 31;
  const int ntn = HID / 16;                   // 64 tiles along N
  const int m0  = (wave / ntn) * 16;          // 4 tiles along M (B=64)
  const int n0  = (wave % ntn) * 16;
  const int l = lane & 15, hi = lane >> 4;

  for (int t = 0; t < SEQ; ++t) {
    const bf16*  Hprev = t ? (Hb16 + (size_t)(t - 1) * BATCH * HID) : h0b;
    const float* Zt    = Z + (size_t)t * BATCH * HID;
    bf16*        Hout  = Hb16 + (size_t)t * BATCH * HID;
    // Warm next step's Z slab into cache while the WMMA K-loop runs
    __builtin_prefetch(Zt + BATCH * HID + (size_t)m0 * HID + n0, 0, 1);

    v8f acc[1];
    wmma_strip<1, HID>(Hprev, HID, Whh, HID, m0, n0, lane, acc);
#pragma unroll
    for (int r = 0; r < 8; ++r) {
      int m = m0 + hi * 8 + r;
      size_t idx = (size_t)m * HID + n0 + l;
      float v = tanhf(acc[0][r] + Zt[idx]);
      Hout[idx] = (bf16)v;                    // feeds next step's A operand
      if (t == SEQ - 1) hF32[idx] = v;        // full-precision h_T
    }

    // ---- grid-wide barrier (monotonic counter: goal = (t+1)*gridDim.x) ----
    __syncthreads();
    if (threadIdx.x == 0) {
      __hip_atomic_fetch_add(barrier, 1u, __ATOMIC_ACQ_REL, __HIP_MEMORY_SCOPE_AGENT);
      const unsigned goal = (unsigned)(t + 1) * gridDim.x;
      while (__hip_atomic_load(barrier, __ATOMIC_ACQUIRE, __HIP_MEMORY_SCOPE_AGENT) < goal)
        __builtin_amdgcn_s_sleep(1);
    }
    __syncthreads();
  }
}

// ----------------------------- Phase 3: out = tanh(H @ W_hx^T + b_hx) -------
// Hb: [SEQ*B, HID] bf16 (row m = t*B+b), Whx: [IN_F, HID] bf16
// Out: [B, SEQ, IN_F] f32
__global__ void gemm_out(const bf16* __restrict__ Hb, const bf16* __restrict__ Whx,
                         const float* __restrict__ b_hx, float* __restrict__ Out) {
  const int wave = (blockIdx.x * blockDim.x + threadIdx.x) >> 5;
  const int lane = threadIdx.x & 31;
  const int nstrips = IN_F / 64;              // 8 strips of 64 along N
  const int m0 = (wave / nstrips) * 16;
  const int n0 = (wave % nstrips) * 64;
  v8f acc[4];
  wmma_strip<4, HID>(Hb, HID, Whx, HID, m0, n0, lane, acc);
  const int l = lane & 15, hi = lane >> 4;
#pragma unroll
  for (int j = 0; j < 4; ++j) {
    const float bias = b_hx[n0 + j * 16 + l];
#pragma unroll
    for (int r = 0; r < 8; ++r) {
      int m = m0 + hi * 8 + r;                // m = t*BATCH + b
      int t = m / BATCH, b = m % BATCH;
      Out[(size_t)b * SEQ * IN_F + (size_t)t * IN_F + n0 + j * 16 + l] =
          tanhf(acc[j][r] + bias);
    }
  }
}

// ----------------------------- Host-side launcher ---------------------------
extern "C" void kernel_launch(void* const* d_in, const int* in_sizes, int n_in,
                              void* d_out, int out_size, void* d_ws, size_t ws_size,
                              hipStream_t stream) {
  const float* x    = (const float*)d_in[0];   // [B, SEQ, IN_F]
  const float* h0   = (const float*)d_in[1];   // [B, HID]
  const float* W_xh = (const float*)d_in[2];   // [HID, IN_F]
  const float* W_hh = (const float*)d_in[3];   // [HID, HID]
  const float* b_hh = (const float*)d_in[4];   // [HID]
  const float* W_hx = (const float*)d_in[5];   // [IN_F, HID]
  const float* b_hx = (const float*)d_in[6];   // [IN_F]
  float* out = (float*)d_out;                  // outs [B,SEQ,IN_F] ++ hT [B,HID]

  // Workspace carve-up (chunk sizes are 256B multiples; base is aligned)
  char* p = (char*)d_ws;
  bf16*  Xb   = (bf16*)p;  p += (size_t)BATCH * SEQ * IN_F * sizeof(bf16);   // 32 MB
  bf16*  Wxhb = (bf16*)p;  p += (size_t)HID * IN_F * sizeof(bf16);           // 1 MB
  bf16*  Whhb = (bf16*)p;  p += (size_t)HID * HID * sizeof(bf16);            // 2 MB
  bf16*  Whxb = (bf16*)p;  p += (size_t)IN_F * HID * sizeof(bf16);           // 1 MB
  bf16*  h0b  = (bf16*)p;  p += (size_t)BATCH * HID * sizeof(bf16);          // 128 KB
  float* Z    = (float*)p; p += (size_t)SEQ * BATCH * HID * sizeof(float);   // 128 MB
  bf16*  Hb16 = (bf16*)p;  p += (size_t)SEQ * BATCH * HID * sizeof(bf16);    // 64 MB
  float* hF32 = (float*)p; p += (size_t)BATCH * HID * sizeof(float);         // 256 KB
  unsigned* barrier = (unsigned*)p;                                          // 4 B

  const int THREADS = 256;
  auto blocks = [](size_t n) { return (unsigned)((n + 255) / 256); };

  // Phase 0: convert inputs/weights to bf16; zero the barrier counter
  init_u32<<<1, 1, 0, stream>>>(barrier);
  cvt_f32_bf16<<<blocks((size_t)BATCH * SEQ * IN_F), THREADS, 0, stream>>>(x, Xb, BATCH * SEQ * IN_F);
  cvt_f32_bf16<<<blocks((size_t)HID * IN_F), THREADS, 0, stream>>>(W_xh, Wxhb, HID * IN_F);
  cvt_f32_bf16<<<blocks((size_t)HID * HID), THREADS, 0, stream>>>(W_hh, Whhb, HID * HID);
  cvt_f32_bf16<<<blocks((size_t)IN_F * HID), THREADS, 0, stream>>>(W_hx, Whxb, IN_F * HID);
  cvt_f32_bf16<<<blocks((size_t)BATCH * HID), THREADS, 0, stream>>>(h0, h0b, BATCH * HID);

  // Phase 1: Z[t,b,:] = x[b,t,:] @ W_xh^T + b_hh (16x64 strip per wave)
  {
    size_t waves = (size_t)(BATCH * SEQ / 16) * (HID / 64);   // 32768
    gemm_xWxh<<<(unsigned)(waves / 8), THREADS, 0, stream>>>(Xb, Wxhb, b_hh, Z);
  }

  // Phase 2: whole recurrence in ONE persistent kernel (32 co-resident blocks,
  // internal grid barrier per step) — removes 511 launch overheads from the
  // serial chain. W_hh (2MB bf16) stays L2-resident across all 512 steps.
  rnn_recurrence<<<(BATCH / 16) * (HID / 16) / 8, THREADS, 0, stream>>>(
      h0b, Whhb, Z, Hb16, hF32, barrier);

  // Phase 3: out[b,t,:] = tanh(H[t,b,:] @ W_hx^T + b_hx) (16x64 strip per wave)
  {
    size_t waves = (size_t)(SEQ * BATCH / 16) * (IN_F / 64);  // 16384
    gemm_out<<<(unsigned)(waves / 8), THREADS, 0, stream>>>(Hb16, Whxb, b_hx, out);
  }

  // Final hidden state h_T (full f32 path) -> tail of d_out
  hipMemcpyAsync(out + (size_t)BATCH * SEQ * IN_F, hF32,
                 (size_t)BATCH * HID * sizeof(float),
                 hipMemcpyDeviceToDevice, stream);
}
